// CoxLoss_36077725286567
// MI455X (gfx1250) — compile-verified, matrix operand order checked by reading, not run on hardware
//
#include <hip/hip_runtime.h>
#include <math.h>
#include <stdint.h>

typedef _Float16 v16h __attribute__((ext_vector_type(16)));
typedef _Float16 half8 __attribute__((ext_vector_type(8)));
typedef float    v8f  __attribute__((ext_vector_type(8)));

#define TILE_J  1024
#define BLOCK   128                              // 4 wave32s
#define WAVES_PER_BLOCK (BLOCK / 32)
#define ROWS_PER_BLOCK  (WAVES_PER_BLOCK * 16)   // 64 rows per block
#define RBLOCK  256                              // reduction block size

// ---------------------------------------------------------------------------
// Kernel 1: T[j] = |y[j]| (f32), w16[j] = exp(theta_j) (f16)
// ---------------------------------------------------------------------------
__global__ void cox_prep(const float* __restrict__ y_hat,
                         const float* __restrict__ y,
                         float* __restrict__ T,
                         _Float16* __restrict__ w16,
                         int n) {
    int i = blockIdx.x * blockDim.x + threadIdx.x;
    if (i < n) {
        T[i]   = fabsf(y[i]);
        w16[i] = (_Float16)expf(y_hat[i]);
    }
}

// ---------------------------------------------------------------------------
// Kernel 2: risk[i] = sum_j exp(theta_j) * (T[j] >= T[i])  via WMMA.
// Each wave32 owns a 16-row i-tile. Per j-chunk of 32:
//   A (16x32 f16) = 0/1 risk-set mask (built from f32 compares, exact),
//   B (32x16 f16) = exp(theta_j) broadcast across N,
//   C (16x16 f32) accumulates risk_sum.
// A and B share identical per-lane K striping on CDNA5:
//   lanes 0-15 : K = {0..7, 16..23},  lanes 16-31 : K = {8..15, 24..31}
// LDS is double-buffered and filled with GLOBAL_LOAD_ASYNC_TO_LDS_B128
// (ASYNCcnt pipeline): issue tile t+1, wait for tile t (in-order completion,
// 3 async instructions per wave per tile), barrier, compute.
// ---------------------------------------------------------------------------
__global__ __launch_bounds__(BLOCK)
void cox_risk_wmma(const float* __restrict__ T,
                   const _Float16* __restrict__ w16,
                   float* __restrict__ risk,
                   int n) {
    __shared__ float    sT[2][TILE_J];           // 2 x 4 KB
    __shared__ _Float16 sW[2][TILE_J];           // 2 x 2 KB

    const int tid  = threadIdx.x;
    const int lane = tid & 31;
    const int wave = tid >> 5;
    const int m    = lane & 15;                  // row within tile / column n
    const int off  = (lane & 16) ? 8 : 0;        // K striping per half-wave
    const int i_base = (blockIdx.x * WAVES_PER_BLOCK + wave) * 16;

    const float Ti = T[i_base + m];
    const int ntiles = n / TILE_J;

    // Async fill of one LDS buffer: each lane moves 3 x 16B.
    auto issue_fill = [&](int buf, int tileBase) {
        const unsigned ldsT = (unsigned)(uintptr_t)&sT[buf][0];
        const unsigned ldsW = (unsigned)(uintptr_t)&sW[buf][0];
        #pragma unroll
        for (int r = 0; r < TILE_J / (BLOCK * 4); ++r) {     // 2 x B128 of T
            unsigned e = (unsigned)(tid + r * BLOCK) * 4u;   // float index
            asm volatile("global_load_async_to_lds_b128 %0, %1, %2"
                         :: "v"(ldsT + e * 4u),
                            "v"((unsigned)(tileBase + (int)e) * 4u),
                            "s"(T)
                         : "memory");
        }
        {                                                     // 1 x B128 of w16
            unsigned e = (unsigned)tid * 8u;                  // half index
            asm volatile("global_load_async_to_lds_b128 %0, %1, %2"
                         :: "v"(ldsW + e * 2u),
                            "v"((unsigned)(tileBase + (int)e) * 2u),
                            "s"(w16)
                         : "memory");
        }
    };

    v8f acc = {0.f, 0.f, 0.f, 0.f, 0.f, 0.f, 0.f, 0.f};

    issue_fill(0, 0);                             // prologue: tile 0 -> buf 0

    for (int t = 0; t < ntiles; ++t) {
        const int buf = t & 1;
        if (t + 1 < ntiles) {
            issue_fill(buf ^ 1, (t + 1) * TILE_J);            // overlap next tile
            asm volatile("s_wait_asynccnt 0x3" ::: "memory"); // tile t complete
        } else {
            asm volatile("s_wait_asynccnt 0x0" ::: "memory");
        }
        __syncthreads();                          // buf[t&1] visible to all waves

        const float*    bT = &sT[buf][0];
        const _Float16* bW = &sW[buf][0];

        for (int jb = 0; jb < TILE_J; jb += 32) {
            const int k0 = jb + off;              // 8-element run 1 (32B aligned)
            const int k1 = jb + 16 + off;         // 8-element run 2

            const float4 ta = *(const float4*)&bT[k0];
            const float4 tb = *(const float4*)&bT[k0 + 4];
            const float4 tc = *(const float4*)&bT[k1];
            const float4 td = *(const float4*)&bT[k1 + 4];
            const half8  w0 = *(const half8*)&bW[k0];
            const half8  w1 = *(const half8*)&bW[k1];

            const float tv[16] = {ta.x, ta.y, ta.z, ta.w, tb.x, tb.y, tb.z, tb.w,
                                  tc.x, tc.y, tc.z, tc.w, td.x, td.y, td.z, td.w};
            v16h A, B;
            #pragma unroll
            for (int h = 0; h < 16; ++h)
                A[h] = (tv[h] >= Ti) ? (_Float16)1.0f : (_Float16)0.0f;
            #pragma unroll
            for (int h = 0; h < 8; ++h) { B[h] = w0[h]; B[8 + h] = w1[h]; }

            acc = __builtin_amdgcn_wmma_f32_16x16x32_f16(
                      /*neg_a=*/false, A, /*neg_b=*/false, B,
                      /*c_mod=*/(short)0, acc,
                      /*reuse_a=*/false, /*reuse_b=*/false);
        }
        __syncthreads();                          // done reading buf before refill
    }

    // C/D layout: VGPR r, lanes 0-15 -> M=r (N=lane); lanes 16-31 -> M=r+8.
    // Column N=0 lives in lane 0 (rows 0-7) and lane 16 (rows 8-15).
    if (m == 0) {
        const int rbase = i_base + ((lane & 16) ? 8 : 0);
        #pragma unroll
        for (int r = 0; r < 8; ++r)
            risk[rbase + r] = acc[r];
    }
}

// ---------------------------------------------------------------------------
// Kernel 3: per-block partial sums of E_i * (theta_i - log(risk_i))
// ---------------------------------------------------------------------------
__global__ void cox_partial(const float* __restrict__ y_hat,
                            const float* __restrict__ y,
                            const float* __restrict__ risk,
                            float* __restrict__ blockSums,
                            int n) {
    __shared__ float red[RBLOCK];
    int i = blockIdx.x * RBLOCK + threadIdx.x;
    float c = 0.f;
    if (i < n && y[i] > 0.f)
        c = y_hat[i] - logf(risk[i]);
    red[threadIdx.x] = c;
    __syncthreads();
    for (int s = RBLOCK / 2; s > 0; s >>= 1) {
        if (threadIdx.x < s) red[threadIdx.x] += red[threadIdx.x + s];
        __syncthreads();
    }
    if (threadIdx.x == 0) blockSums[blockIdx.x] = red[0];
}

// ---------------------------------------------------------------------------
// Kernel 4: final deterministic reduction, loss = -sum / N
// ---------------------------------------------------------------------------
__global__ void cox_final(const float* __restrict__ blockSums,
                          float* __restrict__ out,
                          int nblocks, float invN) {
    __shared__ float red[64];
    float v = (threadIdx.x < nblocks) ? blockSums[threadIdx.x] : 0.f;
    red[threadIdx.x] = v;
    __syncthreads();
    for (int s = 32; s > 0; s >>= 1) {
        if (threadIdx.x < s) red[threadIdx.x] += red[threadIdx.x + s];
        __syncthreads();
    }
    if (threadIdx.x == 0) out[0] = -red[0] * invN;
}

// ---------------------------------------------------------------------------
extern "C" void kernel_launch(void* const* d_in, const int* in_sizes, int n_in,
                              void* d_out, int out_size, void* d_ws, size_t ws_size,
                              hipStream_t stream) {
    const int n = in_sizes[0];                         // 16384
    const float* y_hat = (const float*)d_in[0];
    const float* y     = (const float*)d_in[1];
    float* out = (float*)d_out;

    // Workspace layout (bytes):
    //   [0,     n*4)   T         (f32)
    //   [n*4,   n*6)   w16       (f16)
    //   [n*6,   n*10)  risk      (f32)
    //   [n*10,  ...)   blockSums (f32)
    char* ws = (char*)d_ws;
    float*    T    = (float*)(ws);
    _Float16* w16  = (_Float16*)(ws + (size_t)n * 4);
    float*    risk = (float*)(ws + (size_t)n * 6);
    float*    bsum = (float*)(ws + (size_t)n * 10);

    const int nredblocks = (n + RBLOCK - 1) / RBLOCK;  // 64

    cox_prep<<<nredblocks, RBLOCK, 0, stream>>>(y_hat, y, T, w16, n);

    const int nblocks = n / ROWS_PER_BLOCK;            // 256
    cox_risk_wmma<<<nblocks, BLOCK, 0, stream>>>(T, w16, risk, n);

    cox_partial<<<nredblocks, RBLOCK, 0, stream>>>(y_hat, y, risk, bsum, n);
    cox_final<<<1, 64, 0, stream>>>(bsum, out, nredblocks, 1.0f / (float)n);
}
